// InContextCorrespondence_20650202759332
// MI455X (gfx1250) — compile-verified, matrix operand order checked by reading, not run on hardware
//
#include <hip/hip_runtime.h>
#include <hip/hip_bf16.h>

typedef __attribute__((ext_vector_type(16))) _Float16 v16h;
typedef __attribute__((ext_vector_type(8)))  float    v8f;

#define BATCH 2
#define CH    1280
#define RES   64
#define HW    4096      // RES*RES
#define HG    512
#define KPAD  5         // (11-1)/2

// ---------------------------------------------------------------------------
// 1) Per-row L2 norm reciprocal: feat is [B, C, HW] (channel-first), thread per
//    (b, n); consecutive threads read consecutive n => coalesced.
// ---------------------------------------------------------------------------
__global__ void rownorm_kernel(const float* __restrict__ feat,
                               float* __restrict__ rnorm) {
  int t = blockIdx.x * blockDim.x + threadIdx.x;
  if (t >= BATCH * HW) return;
  int b = t / HW, n = t % HW;
  const float* p = feat + (size_t)b * CH * HW + n;
  float s = 0.f;
#pragma unroll 4
  for (int c = 0; c < CH; ++c) {
    float v = p[(size_t)c * HW];
    s += v * v;
  }
  rnorm[t] = 1.0f / fmaxf(sqrtf(s), 1e-12f);
}

// ---------------------------------------------------------------------------
// 2) Transpose [B,C,HW] -> [B,HW,C] while scaling by rnorm, output f16.
//    32x32 LDS tile transpose, block 32x8.
// ---------------------------------------------------------------------------
__global__ void transpose_scale_kernel(const float* __restrict__ feat,
                                       const float* __restrict__ rnorm,
                                       _Float16* __restrict__ out) {
  __shared__ float tile[32][33];
  int n0 = blockIdx.x * 32, c0 = blockIdx.y * 32, b = blockIdx.z;
  int tx = threadIdx.x, ty = threadIdx.y;
  const float* fp = feat + (size_t)b * CH * HW;
#pragma unroll
  for (int i = 0; i < 4; ++i) {
    int cl = ty + i * 8;
    tile[cl][tx] = fp[(size_t)(c0 + cl) * HW + n0 + tx];
  }
  __syncthreads();
  _Float16* op = out + (size_t)b * HW * CH;
#pragma unroll
  for (int i = 0; i < 4; ++i) {
    int nl = ty + i * 8;
    float r = rnorm[b * HW + n0 + nl];
    op[(size_t)(n0 + nl) * CH + c0 + tx] = (_Float16)(tile[tx][nl] * r);
  }
}

// ---------------------------------------------------------------------------
// 3) Fused GEMM + argmax:  index[b,m] = argmax_n  srcN[b,m,:] . refN[b,n,:]
//    128 threads (4 waves). Block stages 64 src rows (64*1280 f16 = 160 KB)
//    into dynamic LDS via CDNA5 async global->LDS DMA (ASYNCcnt path); each
//    wave owns a 16-row M tile. Loop over 256 N tiles, K in 40 steps of 32 via
//    v_wmma_f32_16x16x32_f16, keep running max/argmax per C register, then
//    butterfly-reduce across the 16-lane column groups.
// ---------------------------------------------------------------------------
__global__ void __launch_bounds__(128)
nn_argmax_kernel(const _Float16* __restrict__ srcN,
                 const _Float16* __restrict__ refN,
                 int* __restrict__ indexOut) {
  extern __shared__ _Float16 sA[];                 // 64 * 1280 halves
  int b      = blockIdx.x >> 6;                    // grid = B * (HW/64)
  int mblock = (blockIdx.x & 63) * 64;

  // Stage the 64-row A block into LDS.
  const _Float16* srcBase = srcN + ((size_t)b * HW + mblock) * CH;
#if defined(__gfx1250__)
  {
    // CDNA5 async copy: global -> LDS without touching VGPR data paths.
    unsigned ldsBase = (unsigned)(size_t)(&sA[0]);   // low 32 bits = LDS offset
    unsigned long long gBase = (unsigned long long)srcBase;
    for (int i = threadIdx.x; i < (64 * CH) / 8; i += 128) {
      unsigned      ldst = ldsBase + (unsigned)i * 16u;
      unsigned long long ga = gBase + (unsigned long long)i * 16ull;
      asm volatile("global_load_async_to_lds_b128 %0, %1, off"
                   :: "v"(ldst), "v"(ga) : "memory");
    }
    asm volatile("s_wait_asynccnt 0" ::: "memory");
  }
#else
  {
    const uint4* gs = (const uint4*)srcBase;
    uint4* ls = (uint4*)sA;
    for (int i = threadIdx.x; i < (64 * CH) / 8; i += 128) ls[i] = gs[i];
  }
#endif
  __syncthreads();

  int lane   = threadIdx.x & 31;
  int wave   = threadIdx.x >> 5;
  int mlocal = wave * 16;
  int nsub   = lane & 15;          // column within tile / row-of-A within tile
  int khalf  = (lane >> 4) * 16;   // K sub-group: lane half h holds k = 16h..16h+15 (contiguous order)

  const _Float16* aRow  = sA + (size_t)(mlocal + nsub) * CH + khalf;
  const _Float16* bBase = refN + (size_t)b * HW * CH + (size_t)nsub * CH + khalf;

  float rmax[8];
  int   ridx[8];
#pragma unroll
  for (int r = 0; r < 8; ++r) { rmax[r] = -3.4e38f; ridx[r] = 0; }

  for (int nt = 0; nt < HW / 16; ++nt) {
    const _Float16* bp = bBase + (size_t)nt * 16 * CH;
    // Warm WGP$/L2 with next N tile while this one computes.
    __builtin_prefetch(bp + 16 * CH, 0, 0);
    v8f c = {};
#pragma unroll 5
    for (int kk = 0; kk < CH; kk += 32) {
      v16h a  = *(const v16h*)(aRow + kk);
      v16h bm = *(const v16h*)(bp + kk);
      c = __builtin_amdgcn_wmma_f32_16x16x32_f16(
          /*neg_a=*/false, a, /*neg_b=*/false, bm,
          /*c_mod=*/(short)0, c, /*reuse_a=*/false, /*reuse_b=*/false);
    }
    int ncol = nt * 16 + nsub;
#pragma unroll
    for (int r = 0; r < 8; ++r) {
      if (c[r] > rmax[r]) { rmax[r] = c[r]; ridx[r] = ncol; }  // strict > keeps first max
    }
  }

  // Reduce across the 16 lanes of each half-wave (columns of the tile),
  // tie-break on lowest index like jnp.argmax.
#pragma unroll
  for (int m = 1; m < 16; m <<= 1) {
#pragma unroll
    for (int r = 0; r < 8; ++r) {
      float om = __shfl_xor(rmax[r], m, 16);
      int   oi = __shfl_xor(ridx[r], m, 16);
      if (om > rmax[r] || (om == rmax[r] && oi < ridx[r])) {
        rmax[r] = om; ridx[r] = oi;
      }
    }
  }
  if (nsub == 0) {
    // C-tile layout: VGPR r -> row r (lanes 0-15) or row r+8 (lanes 16-31)
    int rowbase = b * HW + mblock + mlocal + ((lane >> 4) ? 8 : 0);
#pragma unroll
    for (int r = 0; r < 8; ++r) indexOut[rowbase + r] = ridx[r];
  }
}

// ---------------------------------------------------------------------------
// 4) Binarize + 8x8 min-pool guidance [B,1,512,512] -> mask_ref [B,4096]
// ---------------------------------------------------------------------------
__global__ void maskpool_kernel(const float* __restrict__ g,
                                float* __restrict__ maskRef) {
  int t = blockIdx.x * blockDim.x + threadIdx.x;
  if (t >= BATCH * HW) return;
  int b = t / HW, n = t % HW;
  int y = (n >> 6) * 8, x = (n & 63) * 8;
  const float* gp = g + (size_t)b * HG * HG;
  float mn = 1e30f;
  for (int dy = 0; dy < 8; ++dy)
    for (int dx = 0; dx < 8; ++dx) {
      float v = gp[(size_t)(y + dy) * HG + (x + dx)];
      v = v > 0.f ? 1.f : v;
      mn = fminf(mn, v);
    }
  maskRef[t] = mn;
}

// ---------------------------------------------------------------------------
// 5) Gather pooled mask through correspondence index
// ---------------------------------------------------------------------------
__global__ void gather_kernel(const float* __restrict__ maskRef,
                              const int* __restrict__ idx,
                              float* __restrict__ maskRes) {
  int t = blockIdx.x * blockDim.x + threadIdx.x;
  if (t >= BATCH * HW) return;
  int b = t / HW;
  maskRes[t] = maskRef[b * HW + idx[t]];
}

// ---------------------------------------------------------------------------
// 6) Bilinear upsample 64 -> 512 (half-pixel, align_corners=False). Writes
//    both the workspace copy (for trimap) and the third output slot.
// ---------------------------------------------------------------------------
__global__ void upsample_kernel(const float* __restrict__ maskRes,
                                float* __restrict__ maskUpWs,
                                float* __restrict__ maskUpOut) {
  int t = blockIdx.x * blockDim.x + threadIdx.x;
  if (t >= BATCH * HG * HG) return;
  int b = t / (HG * HG);
  int p = t % (HG * HG);
  int y = p / HG, x = p % HG;
  float fy = (y + 0.5f) * ((float)RES / HG) - 0.5f;
  float fx = (x + 0.5f) * ((float)RES / HG) - 0.5f;
  int y0 = (int)floorf(fy), x0 = (int)floorf(fx);
  float wy = fy - (float)y0, wx = fx - (float)x0;
  int y0c = min(max(y0, 0), RES - 1), y1c = min(max(y0 + 1, 0), RES - 1);
  int x0c = min(max(x0, 0), RES - 1), x1c = min(max(x0 + 1, 0), RES - 1);
  const float* mp = maskRes + (size_t)b * HW;
  float v00 = mp[y0c * RES + x0c], v01 = mp[y0c * RES + x1c];
  float v10 = mp[y1c * RES + x0c], v11 = mp[y1c * RES + x1c];
  float v = (1.f - wy) * ((1.f - wx) * v00 + wx * v01) +
            wy * ((1.f - wx) * v10 + wx * v11);
  maskUpWs[t]  = v;
  maskUpOut[t] = v;
}

// ---------------------------------------------------------------------------
// 7) Trimap: reflect-padded 11x11 max (dilate) and min (erode == 1-dilate(1-x))
// ---------------------------------------------------------------------------
__device__ __forceinline__ int refl(int i) {
  if (i < 0) i = -i;
  if (i >= HG) i = 2 * (HG - 1) - i;
  return i;
}

__global__ void trimap_kernel(const float* __restrict__ maskUp,
                              float* __restrict__ trimapOut) {
  int t = blockIdx.x * blockDim.x + threadIdx.x;
  if (t >= BATCH * HG * HG) return;
  int b = t / (HG * HG);
  int p = t % (HG * HG);
  int y = p / HG, x = p % HG;
  const float* mp = maskUp + (size_t)b * HG * HG;
  float mx = -1e30f, mn = 1e30f;
  for (int dy = -KPAD; dy <= KPAD; ++dy) {
    int yy = refl(y + dy);
    const float* row = mp + (size_t)yy * HG;
    for (int dx = -KPAD; dx <= KPAD; ++dx) {
      float v = row[refl(x + dx)];
      mx = fmaxf(mx, v);
      mn = fminf(mn, v);
    }
  }
  float tri = (mx == 1.0f) ? 0.5f : 0.0f;   // dilated == 1
  if (mn == 1.0f) tri = 1.0f;               // eroded  == 1
  trimapOut[t] = tri;
}

// ---------------------------------------------------------------------------
extern "C" void kernel_launch(void* const* d_in, const int* in_sizes, int n_in,
                              void* d_out, int out_size, void* d_ws, size_t ws_size,
                              hipStream_t stream) {
  const float* ref_f = (const float*)d_in[0];   // feature_of_reference_image
  const float* src_f = (const float*)d_in[1];   // ft_source
  // d_in[2] = attn_source: unused by the reference
  const float* guid  = (const float*)d_in[3];   // guidance_on_reference_image

  char* ws = (char*)d_ws;
  size_t off = 0;
  _Float16* srcN  = (_Float16*)(ws + off); off += (size_t)BATCH * HW * CH * sizeof(_Float16);
  _Float16* refN  = (_Float16*)(ws + off); off += (size_t)BATCH * HW * CH * sizeof(_Float16);
  float* rnormS   = (float*)(ws + off);    off += (size_t)BATCH * HW * sizeof(float);
  float* rnormR   = (float*)(ws + off);    off += (size_t)BATCH * HW * sizeof(float);
  int*   idx      = (int*)(ws + off);      off += (size_t)BATCH * HW * sizeof(int);
  float* maskRef  = (float*)(ws + off);    off += (size_t)BATCH * HW * sizeof(float);
  float* maskRes  = (float*)(ws + off);    off += (size_t)BATCH * HW * sizeof(float);
  float* maskUp   = (float*)(ws + off);    off += (size_t)BATCH * HG * HG * sizeof(float);

  // Output layout: trimap [B,1,512,512] | ft_source [B,C,64,64] | mask_up [B,1,512,512]
  float* outTrimap = (float*)d_out;
  float* outFt     = outTrimap + (size_t)BATCH * HG * HG;
  float* outMaskUp = outFt + (size_t)BATCH * CH * HW;

  const int nRows = BATCH * HW;

  rownorm_kernel<<<(nRows + 255) / 256, 256, 0, stream>>>(src_f, rnormS);
  rownorm_kernel<<<(nRows + 255) / 256, 256, 0, stream>>>(ref_f, rnormR);

  dim3 tgrid(HW / 32, CH / 32, BATCH), tblk(32, 8);
  transpose_scale_kernel<<<tgrid, tblk, 0, stream>>>(src_f, rnormS, srcN);
  transpose_scale_kernel<<<tgrid, tblk, 0, stream>>>(ref_f, rnormR, refN);

  // 128 blocks x 128 threads, 160 KB dynamic LDS per block (A block stage)
  nn_argmax_kernel<<<BATCH * (HW / 64), 128, 64 * CH * sizeof(_Float16), stream>>>(
      srcN, refN, idx);

  maskpool_kernel<<<(nRows + 255) / 256, 256, 0, stream>>>(guid, maskRef);
  gather_kernel<<<(nRows + 255) / 256, 256, 0, stream>>>(maskRef, idx, maskRes);

  const int nPix = BATCH * HG * HG;
  upsample_kernel<<<(nPix + 255) / 256, 256, 0, stream>>>(maskRes, maskUp, outMaskUp);
  trimap_kernel<<<(nPix + 255) / 256, 256, 0, stream>>>(maskUp, outTrimap);

  // ft_source passthrough
  hipMemcpyAsync(outFt, src_f, (size_t)BATCH * CH * HW * sizeof(float),
                 hipMemcpyDeviceToDevice, stream);
}
